// RelPartialLearnableMultiHeadAttn_84533546320101
// MI455X (gfx1250) — compile-verified, matrix operand order checked by reading, not run on hardware
//
#include <hip/hip_runtime.h>
#include <hip/hip_bf16.h>
#include <math.h>

// ---------------------------------------------------------------------------
// Transformer-XL RelPartialLearnableMultiHeadAttn forward for MI455X (gfx1250)
// bf16 WMMA (v_wmma_f32_16x16x32_bf16) everywhere, fp32 accumulate.
// ---------------------------------------------------------------------------

typedef __attribute__((ext_vector_type(16))) __bf16 v16bf;
typedef __attribute__((ext_vector_type(8)))  float  v8f;
typedef __attribute__((ext_vector_type(16))) float  v16f;

constexpr int NH   = 16;
constexpr int DH   = 64;
constexpr int DM   = 1024;
constexpr int QLEN = 1024;
constexpr int MLEN = 1024;
constexpr int BSZ  = 2;
constexpr int KLEN = QLEN + MLEN;   // 2048
constexpr int HD3  = 3 * NH * DH;   // 3072

// ---------------------------------------------------------------------------
// Elementwise fp32 -> bf16 conversion
// ---------------------------------------------------------------------------
__global__ void cvt_f32_bf16(const float* __restrict__ src,
                             __bf16* __restrict__ dst, long n) {
  long i = (long)blockIdx.x * blockDim.x + threadIdx.x;
  if (i < n) dst[i] = (__bf16)src[i];
}

// cat = concat(mems, w) along sequence axis; flat layouts line up exactly.
__global__ void build_cat_bf16(const float* __restrict__ mems,
                               const float* __restrict__ w,
                               __bf16* __restrict__ cat) {
  long i = (long)blockIdx.x * blockDim.x + threadIdx.x;
  constexpr long NMEM = (long)MLEN * BSZ * DM;   // 2097152
  constexpr long NTOT = (long)KLEN * BSZ * DM;   // 4194304
  if (i < NTOT) {
    float v = (i < NMEM) ? mems[i] : w[i - NMEM];
    cat[i] = (__bf16)v;
  }
}

// ---------------------------------------------------------------------------
// Generic GEMM: C[M,N] = A[M,K] * B[N,K]^T  (A,B bf16 row-major in K)
// One wave per block computes a 16(M) x 64(N) tile with 4 fp32 accumulators.
// ---------------------------------------------------------------------------
template <bool F32OUT>
__global__ void gemm_bf16(const __bf16* __restrict__ A,
                          const __bf16* __restrict__ B,
                          void* __restrict__ Cv, int M, int N, int K) {
  int lane = threadIdx.x;
  int col  = lane & 15;     // M row (A) / N col (B,C)
  int hv   = lane >> 4;     // K half-chunk select
  long nbase = (long)blockIdx.x * 64;
  long mbase = (long)blockIdx.y * 16;

  v8f acc[4] = {};
  const __bf16* arow = A + (mbase + col) * (long)K + hv * 16;

  for (int kk = 0; kk < K; kk += 32) {
    v16bf a = *(const v16bf*)(arow + kk);
#pragma unroll
    for (int j = 0; j < 4; ++j) {
      const __bf16* brow = B + (nbase + j * 16 + col) * (long)K + kk + hv * 16;
      v16bf bf = *(const v16bf*)brow;
      acc[j] = __builtin_amdgcn_wmma_f32_16x16x32_bf16(
          false, a, false, bf, (short)0, acc[j], false, false);
    }
  }
#pragma unroll
  for (int j = 0; j < 4; ++j) {
#pragma unroll
    for (int i = 0; i < 8; ++i) {
      long m = mbase + i + hv * 8;
      long n = nbase + j * 16 + col;
      float v = acc[j][i];
      if (F32OUT) ((float*)Cv)[m * N + n] = v;
      else        ((__bf16*)Cv)[m * N + n] = (__bf16)v;
    }
  }
}

// ---------------------------------------------------------------------------
// AC scores: score[b,n,q,k] = (q[q,:]+r_w_bias[n,:]) . k[k,:]   (unscaled)
// heads layout: [(s*BSZ+b)][3072] ; q cols [0,1024) k cols [1024,2048)
// grid: x = KLEN/64 k-tiles, y = QLEN/16 q-tiles, z = b*NH+n ; block = 32
// ---------------------------------------------------------------------------
__global__ void attn_ac(const __bf16* __restrict__ heads,
                        const float* __restrict__ r_w_bias,
                        float* __restrict__ score) {
  int lane = threadIdx.x;
  int col = lane & 15, hv = lane >> 4;
  int kbase = blockIdx.x * 64;
  int qbase = blockIdx.y * 16;
  int z = blockIdx.z, b = z >> 4, n = z & 15;

  // A fragments: q row + r_w_bias, converted back to bf16 (2 K-chunks of 32)
  v16bf afr[2];
  long qrow = ((long)(MLEN + qbase + col) * BSZ + b) * HD3 + n * DH;
#pragma unroll
  for (int c = 0; c < 2; ++c) {
    int kd = c * 32 + hv * 16;
    v16bf qa = *(const v16bf*)(heads + qrow + kd);
#pragma unroll
    for (int t = 0; t < 16; ++t)
      afr[c][t] = (__bf16)((float)qa[t] + r_w_bias[n * DH + kd + t]);
  }

  v8f acc[4] = {};
#pragma unroll
  for (int c = 0; c < 2; ++c) {
#pragma unroll
    for (int j = 0; j < 4; ++j) {
      long krow = ((long)(kbase + j * 16 + col) * BSZ + b) * HD3
                  + NH * DH + n * DH + c * 32 + hv * 16;
      v16bf bf = *(const v16bf*)(heads + krow);
      acc[j] = __builtin_amdgcn_wmma_f32_16x16x32_bf16(
          false, afr[c], false, bf, (short)0, acc[j], false, false);
    }
  }
#pragma unroll
  for (int j = 0; j < 4; ++j) {
#pragma unroll
    for (int i = 0; i < 8; ++i) {
      long q = qbase + i + hv * 8;
      score[((long)z * QLEN + q) * KLEN + kbase + j * 16 + col] = acc[j][i];
    }
  }
}

// ---------------------------------------------------------------------------
// BD scores with fused rel-shift:
//   BD_raw[q,p] = (q[q,:]+r_r_bias[n,:]) . rk[p,n,:]
//   scatter-add into score[q, j] at j = p + q - 1023 (valid j only).
// rk stored row-major [p][NH*DH] (raw R-projection GEMM output).
// ---------------------------------------------------------------------------
__global__ void attn_bd(const __bf16* __restrict__ heads,
                        const __bf16* __restrict__ rk,
                        const float* __restrict__ r_r_bias,
                        float* __restrict__ score) {
  int pbase = blockIdx.x * 64;
  int qbase = blockIdx.y * 16;
  // uniform tile-level culling (max j < 0, or min j >= KLEN)
  if (pbase + qbase + 78 < 1023) return;
  if (pbase + qbase - 1023 >= KLEN) return;

  int lane = threadIdx.x;
  int col = lane & 15, hv = lane >> 4;
  int z = blockIdx.z, b = z >> 4, n = z & 15;

  v16bf afr[2];
  long qrow = ((long)(MLEN + qbase + col) * BSZ + b) * HD3 + n * DH;
#pragma unroll
  for (int c = 0; c < 2; ++c) {
    int kd = c * 32 + hv * 16;
    v16bf qa = *(const v16bf*)(heads + qrow + kd);
#pragma unroll
    for (int t = 0; t < 16; ++t)
      afr[c][t] = (__bf16)((float)qa[t] + r_r_bias[n * DH + kd + t]);
  }

  v8f acc[4] = {};
#pragma unroll
  for (int c = 0; c < 2; ++c) {
#pragma unroll
    for (int j = 0; j < 4; ++j) {
      long prow = (long)(pbase + j * 16 + col) * (NH * DH)
                  + n * DH + c * 32 + hv * 16;
      v16bf bf = *(const v16bf*)(rk + prow);
      acc[j] = __builtin_amdgcn_wmma_f32_16x16x32_bf16(
          false, afr[c], false, bf, (short)0, acc[j], false, false);
    }
  }
#pragma unroll
  for (int j = 0; j < 4; ++j) {
#pragma unroll
    for (int i = 0; i < 8; ++i) {
      int q  = qbase + i + hv * 8;
      int p  = pbase + j * 16 + col;
      int jj = p + q - 1023;                 // rel-shifted key index
      if (jj >= 0 && jj < KLEN) {
        long idx = ((long)z * QLEN + q) * KLEN + jj;
        score[idx] += acc[j][i];
      }
    }
  }
}

// ---------------------------------------------------------------------------
// In-place masked softmax over each score row (KLEN=2048). scale = 1/sqrt(DH).
// One 256-thread block per row; mask j > q + MLEN -> -inf.
// ---------------------------------------------------------------------------
__global__ void softmax_rows(float* __restrict__ score) {
  __shared__ float red[256];
  long row = blockIdx.x;                 // (b*NH+n)*QLEN + q
  int q = (int)(row & (QLEN - 1));
  long base = row * KLEN;
  int t = threadIdx.x;

  float vals[8];
  float mx = -INFINITY;
#pragma unroll
  for (int i = 0; i < 8; ++i) {
    int j = t + i * 256;
    float s = score[base + j] * 0.125f;  // 1/sqrt(64)
    if (j > q + MLEN) s = -INFINITY;
    vals[i] = s;
    mx = fmaxf(mx, s);
  }
  red[t] = mx;
  __syncthreads();
  for (int o = 128; o > 0; o >>= 1) {
    if (t < o) red[t] = fmaxf(red[t], red[t + o]);
    __syncthreads();
  }
  float rowmax = red[0];
  __syncthreads();

  float sum = 0.f;
#pragma unroll
  for (int i = 0; i < 8; ++i) {
    float e = __expf(vals[i] - rowmax);  // -inf -> 0
    vals[i] = e;
    sum += e;
  }
  red[t] = sum;
  __syncthreads();
  for (int o = 128; o > 0; o >>= 1) {
    if (t < o) red[t] += red[t + o];
    __syncthreads();
  }
  float inv = 1.0f / red[0];
#pragma unroll
  for (int i = 0; i < 8; ++i)
    score[base + t + i * 256] = vals[i] * inv;
}

// ---------------------------------------------------------------------------
// V transpose: vT[(b*NH+n)*DH + d][k] = heads[(k*BSZ+b)*HD3 + 2048 + n*DH + d]
// 64x64 tiles staged through LDS (padded) for coalesced read+write.
// ---------------------------------------------------------------------------
__global__ void transpose_v(const __bf16* __restrict__ heads,
                            __bf16* __restrict__ vT) {
  __shared__ __bf16 tile[64][65];
  int z = blockIdx.z, b = z >> 4, n = z & 15;
  int kbase = blockIdx.x * 64;
  int t = threadIdx.x;
#pragma unroll
  for (int r = 0; r < 16; ++r) {
    int idx = r * 256 + t;
    int kl = idx >> 6, dl = idx & 63;
    tile[kl][dl] = heads[((long)(kbase + kl) * BSZ + b) * HD3
                         + 2 * NH * DH + n * DH + dl];
  }
  __syncthreads();
#pragma unroll
  for (int r = 0; r < 16; ++r) {
    int idx = r * 256 + t;
    int dl = idx >> 6, kl = idx & 63;
    vT[((long)z * DH + dl) * KLEN + kbase + kl] = tile[kl][dl];
  }
}

// ---------------------------------------------------------------------------
// PV GEMM: attn_vec[q,d] = sum_k prob[q,k] * v[k,d]
// A = prob row (fp32, converted to bf16 on load), B = vT rows (d on lanes).
// Output layout [(q*BSZ+b)][NH*DH] for the final projection GEMM.
// ---------------------------------------------------------------------------
__global__ void attn_pv(const float* __restrict__ prob,
                        const __bf16* __restrict__ vT,
                        __bf16* __restrict__ av) {
  int lane = threadIdx.x;
  int col = lane & 15, hv = lane >> 4;
  int qbase = blockIdx.y * 16;
  int z = blockIdx.z, b = z >> 4, n = z & 15;

  v8f acc[4] = {};
  long prow = ((long)z * QLEN + qbase + col) * KLEN + hv * 16;
  long vbase = (long)z * DH * KLEN;

  for (int kk = 0; kk < KLEN; kk += 32) {
    v16f pf = *(const v16f*)(prob + prow + kk);
    v16bf a;
#pragma unroll
    for (int t = 0; t < 16; ++t) a[t] = (__bf16)pf[t];
#pragma unroll
    for (int j = 0; j < 4; ++j) {
      const __bf16* bp = vT + vbase + (long)(j * 16 + col) * KLEN + kk + hv * 16;
      v16bf bf = *(const v16bf*)bp;
      acc[j] = __builtin_amdgcn_wmma_f32_16x16x32_bf16(
          false, a, false, bf, (short)0, acc[j], false, false);
    }
  }
#pragma unroll
  for (int j = 0; j < 4; ++j) {
#pragma unroll
    for (int i = 0; i < 8; ++i) {
      long q = qbase + i + hv * 8;
      av[(q * BSZ + b) * (long)(NH * DH) + n * DH + j * 16 + col] =
          (__bf16)acc[j][i];
    }
  }
}

// ---------------------------------------------------------------------------
// Host launcher
// ---------------------------------------------------------------------------
extern "C" void kernel_launch(void* const* d_in, const int* in_sizes, int n_in,
                              void* d_out, int out_size, void* d_ws, size_t ws_size,
                              hipStream_t stream) {
  const float* w    = (const float*)d_in[0];
  const float* r    = (const float*)d_in[1];
  const float* rwb  = (const float*)d_in[2];
  const float* rrb  = (const float*)d_in[3];
  const float* mems = (const float*)d_in[4];
  const float* Wqkv = (const float*)d_in[5];
  const float* Wr   = (const float*)d_in[6];
  const float* Wo   = (const float*)d_in[7];
  // d_in[8] attn_mask is recomputed analytically (j > q + MLEN)

  float* out  = (float*)d_out;
  float* prob = out + (long)QLEN * BSZ * DM;   // score/prob region, in-place

  char* ws = (char*)d_ws;
  __bf16* catb   = (__bf16*)(ws);                   //  8 MB: 4096x1024
  __bf16* wqkvb  = (__bf16*)(ws + ( 8l << 20));     //  6 MB: 3072x1024
  __bf16* wrb    = (__bf16*)(ws + (14l << 20));     //  2 MB: 1024x1024
  __bf16* wob    = (__bf16*)(ws + (16l << 20));     //  2 MB: 1024x1024
  __bf16* rb     = (__bf16*)(ws + (18l << 20));     //  4 MB: 2048x1024
  __bf16* headsb = (__bf16*)(ws + (22l << 20));     // 24 MB: 4096x3072
  __bf16* rkb    = (__bf16*)(ws + (46l << 20));     //  4 MB: 2048x1024
  __bf16* vTb    = (__bf16*)(ws + (50l << 20));     //  8 MB: 32x64x2048
  __bf16* avb    = (__bf16*)(ws + (58l << 20));     //  4 MB: 2048x1024

  // 1) fp32 -> bf16 conversions
  {
    long n = (long)KLEN * BSZ * DM;  // cat
    build_cat_bf16<<<(n + 255) / 256, 256, 0, stream>>>(mems, w, catb);
  }
  cvt_f32_bf16<<<((long)HD3 * DM + 255) / 256, 256, 0, stream>>>(Wqkv, wqkvb, (long)HD3 * DM);
  cvt_f32_bf16<<<((long)DM  * DM + 255) / 256, 256, 0, stream>>>(Wr,   wrb,   (long)DM * DM);
  cvt_f32_bf16<<<((long)DM  * DM + 255) / 256, 256, 0, stream>>>(Wo,   wob,   (long)DM * DM);
  cvt_f32_bf16<<<((long)KLEN* DM + 255) / 256, 256, 0, stream>>>(r,    rb,    (long)KLEN * DM);

  // 2) QKV projection: heads[4096,3072] = cat @ Wqkv^T
  gemm_bf16<false><<<dim3(HD3 / 64, (KLEN * BSZ) / 16), 32, 0, stream>>>(
      catb, wqkvb, headsb, KLEN * BSZ, HD3, DM);

  // 3) R projection: rk[2048,1024] = r @ Wr^T
  gemm_bf16<false><<<dim3(DM / 64, KLEN / 16), 32, 0, stream>>>(
      rb, wrb, rkb, KLEN, DM, DM);

  // 4) V transpose into [b,n][d][k]
  transpose_v<<<dim3(KLEN / 64, 1, BSZ * NH), 256, 0, stream>>>(headsb, vTb);

  // 5) AC scores (unscaled) into prob region
  attn_ac<<<dim3(KLEN / 64, QLEN / 16, BSZ * NH), 32, 0, stream>>>(
      headsb, rwb, prob);

  // 6) BD scores with fused rel-shift, scatter-added
  attn_bd<<<dim3(KLEN / 64, QLEN / 16, BSZ * NH), 32, 0, stream>>>(
      headsb, rkb, rrb, prob);

  // 7) masked softmax in place -> prob (second output)
  softmax_rows<<<(long)BSZ * NH * QLEN, 256, 0, stream>>>(prob);

  // 8) attn_vec = prob @ v
  attn_pv<<<dim3(1, QLEN / 16, BSZ * NH), 32, 0, stream>>>(prob, vTb, avb);

  // 9) out[2048,1024] = attn_vec @ Wo^T (fp32 into d_out)
  gemm_bf16<true><<<dim3(DM / 64, (QLEN * BSZ) / 16), 32, 0, stream>>>(
      avb, wob, (void*)out, QLEN * BSZ, DM, DM);
}
